// GAT_1829656068113
// MI455X (gfx1250) — matrix operand order repaired and checked by reading, not compiled, not run on hardware
//
#include <hip/hip_runtime.h>
#include <hip/hip_bf16.h>
#include <math.h>

// ---------------------------------------------------------------------------
// Types for CDNA5 WMMA (wave32): bf16 A/B fragments, f32 accumulator.
// ---------------------------------------------------------------------------
typedef __attribute__((ext_vector_type(4)))  int    v4i;
typedef __attribute__((ext_vector_type(4)))  float  v4f;
typedef __attribute__((ext_vector_type(8)))  float  v8f;
typedef __attribute__((ext_vector_type(8)))  __bf16 v8bf;
typedef __attribute__((ext_vector_type(16))) __bf16 v16bf;

// Address-space-qualified int4 for the async-copy builtin signature:
//   void __builtin_amdgcn_global_load_async_to_lds_b128(
//       v4i addrspace(1)* gsrc, v4i addrspace(3)* ldst, imm offset, imm cpol)
typedef v4i __attribute__((address_space(1))) v4i_g;
typedef v4i __attribute__((address_space(3))) v4i_l;

// ---------------------------------------------------------------------------
// CDNA5 async global->LDS copy (ASYNCcnt path). Guarded so compile always
// succeeds: falls back to a direct LDS store if the builtin is unavailable.
// ---------------------------------------------------------------------------
#if defined(__AMDGCN__) && __has_builtin(__builtin_amdgcn_global_load_async_to_lds_b128)
  #define HAVE_ASYNC 1
#else
  #define HAVE_ASYNC 0
#endif

__device__ __forceinline__ void cp_async16(void* lds, const void* g) {
#if HAVE_ASYNC
    __builtin_amdgcn_global_load_async_to_lds_b128(
        (v4i_g*)g, (v4i_l*)lds, 0, 0);
#else
    *(v4f*)lds = *(const v4f*)g;
#endif
}

#if HAVE_ASYNC
  #if __has_builtin(__builtin_amdgcn_s_wait_asynccnt)
    #define WAIT_ASYNC(n) __builtin_amdgcn_s_wait_asynccnt(n)
  #else
    #define WAIT_ASYNC(n) asm volatile("s_wait_asynccnt %0" :: "i"(n) : "memory")
  #endif
#else
  #define WAIT_ASYNC(n) ((void)0)
#endif

__device__ inline float elu1(float x) { return x > 0.f ? x : (expf(x) - 1.f); }

// float atomic max via sign-aware integer punning (monotone bit trick)
__device__ inline void atomicMaxF(float* addr, float val) {
    if (val >= 0.f) atomicMax((int*)addr, __float_as_int(val));
    else            atomicMin((unsigned int*)addr, __float_as_uint(val));
}

// Build a 16-bit-A WMMA fragment from an fp32 LDS row: lanes 0-15 carry
// K kk+0..15, lanes 16-31 carry K kk+16..31.
__device__ __forceinline__ v16bf afrag_f32(const float* ap, int kb, int kk) {
    v16bf r;
    #pragma unroll
    for (int j = 0; j < 8; ++j) {
        r[j]     = (__bf16)ap[kk + kb * 16 + j];
        r[8 + j] = (__bf16)ap[kk + kb * 16 + 8 + j];
    }
    return r;
}

// ---------------------------------------------------------------------------
// Kernel 0: one-shot weight prep — Wt[col*K + k] = bf16(W[k*64 + col])
// (transposed so per-block B tiles are contiguous and async-copyable)
// ---------------------------------------------------------------------------
__global__ void k_cvtW(const float* __restrict__ W, __bf16* __restrict__ Wt, int K)
{
    int t = blockIdx.x * blockDim.x + threadIdx.x;
    if (t >= K * 64) return;
    int col = t / K, k = t - col * K;
    Wt[t] = (__bf16)W[(size_t)k * 64 + col];
}

// ---------------------------------------------------------------------------
// Kernel 1: h1 = ELU(BN1(x @ proj_W + proj_b))
//   Double-buffered async staging: A tile fp32 (128x32, raw bytes -> LDS),
//   B tile bf16 (pre-transposed Wt). 8 waves x (16 rows x 64 cols) strip,
//   4x v_wmma_f32_16x16x32_bf16 per K-step, 24 K-steps.
// ---------------------------------------------------------------------------
__global__ __launch_bounds__(256) void k_proj(
    const float* __restrict__ x, const __bf16* __restrict__ Wt,
    const float* __restrict__ bias,
    const float* __restrict__ bn_g, const float* __restrict__ bn_b,
    const float* __restrict__ bn_m, const float* __restrict__ bn_v,
    float* __restrict__ out, int n)
{
    __shared__ alignas(128) float  Af[2][128 * 32];  // 2 x 16KB fp32 A tiles
    __shared__ alignas(64)  __bf16 Bs[2][64 * 32];   // 2 x 4KB bf16 B^T tiles
    const int tid  = threadIdx.x;
    const int wave = tid >> 5;
    const int lane = tid & 31;
    const int rowBase = blockIdx.x * 128;

    // Issue one stage: 4 A-chunks + 1 B-chunk per thread = 5 async ops/wave.
    auto stage = [&](int buf, int k0) {
        #pragma unroll
        for (int c = 0; c < 4; ++c) {
            int i4 = tid + c * 256;              // float4 index in [0,1024)
            int r = i4 >> 3, cc = (i4 & 7) * 4;
            int gr = rowBase + r; if (gr >= n) gr = n - 1;   // clamp (rows guarded at store)
            cp_async16(&Af[buf][r * 32 + cc], &x[(size_t)gr * 768 + k0 + cc]);
        }
        {
            int col = tid >> 2, part = (tid & 3) * 8;        // 256 x 16B chunks
            cp_async16(&Bs[buf][col * 32 + part], &Wt[(size_t)col * 768 + k0 + part]);
        }
    };

    v8f acc[4] = {};
    stage(0, 0);
    int buf = 0;
    for (int s = 0; s < 24; ++s) {
        if (s + 1 < 24) {
            stage(buf ^ 1, (s + 1) * 32);
            WAIT_ASYNC(5);   // 5 newest (next stage) may remain; current stage done
        } else {
            WAIT_ASYNC(0);
        }
        __syncthreads();

        const float* ap = &Af[buf][(wave * 16 + (lane & 15)) * 32];
        const int kb = lane >> 4;
        v16bf afrag = afrag_f32(ap, kb, 0);
        #pragma unroll
        for (int t = 0; t < 4; ++t) {
            const __bf16* bp = &Bs[buf][(t * 16 + (lane & 15)) * 32];
            v16bf bfrag = *(const v16bf*)(bp + kb * 16);
            acc[t] = __builtin_amdgcn_wmma_f32_16x16x32_bf16(
                false, afrag, false, bfrag, (short)0, acc[t], false, false);
        }
        __syncthreads();     // readers done before this buffer is overwritten
        buf ^= 1;
    }

    // Epilogue: bias + BN + ELU. C/D layout: vgpr j -> row j (lanes 0-15) or 8+j.
    #pragma unroll
    for (int t = 0; t < 4; ++t) {
        int col = t * 16 + (lane & 15);
        float rs = rsqrtf(bn_v[col] + 1e-5f);
        float gg = bn_g[col], bb = bn_b[col], mm = bn_m[col], bi = bias[col];
        #pragma unroll
        for (int j = 0; j < 8; ++j) {
            int row = rowBase + wave * 16 + (lane >> 4) * 8 + j;
            if (row < n) {
                float h = acc[t][j] + bi;
                h = (h - mm) * rs * gg + bb;
                out[(size_t)row * 64 + col] = elu1(h);
            }
        }
    }
}

// ---------------------------------------------------------------------------
// Kernel 2: xw = in[N,64] @ W[64,64] via pre-transposed bf16 Wt (async staged)
// ---------------------------------------------------------------------------
__global__ __launch_bounds__(256) void k_gemm64(
    const float* __restrict__ in, const __bf16* __restrict__ Wt,
    float* __restrict__ out, int n)
{
    __shared__ alignas(128) float  Af[128 * 64];   // 32KB fp32 A tile
    __shared__ alignas(64)  __bf16 Bs[64 * 64];    // 8KB  bf16 B^T
    const int tid  = threadIdx.x;
    const int wave = tid >> 5;
    const int lane = tid & 31;
    const int rowBase = blockIdx.x * 128;

    #pragma unroll
    for (int c = 0; c < 8; ++c) {                  // A: 2048 float4 chunks
        int i4 = tid + c * 256;
        int r = i4 >> 4, cc = (i4 & 15) * 4;
        int gr = rowBase + r; if (gr >= n) gr = n - 1;
        cp_async16(&Af[r * 64 + cc], &in[(size_t)gr * 64 + cc]);
    }
    #pragma unroll
    for (int c = 0; c < 2; ++c) {                  // B: 512 x 16B chunks
        int i4 = tid + c * 256;
        int col = i4 >> 3, part = (i4 & 7) * 8;
        cp_async16(&Bs[col * 64 + part], &Wt[(size_t)col * 64 + part]);
    }
    WAIT_ASYNC(0);
    __syncthreads();

    v8f acc[4] = {};
    const float* ap = &Af[(wave * 16 + (lane & 15)) * 64];
    const int kb = lane >> 4;
    #pragma unroll
    for (int kk = 0; kk < 64; kk += 32) {
        v16bf afrag = afrag_f32(ap, kb, kk);
        #pragma unroll
        for (int t = 0; t < 4; ++t) {
            const __bf16* bp = &Bs[(t * 16 + (lane & 15)) * 64];
            v16bf bfrag = *(const v16bf*)(bp + kk + kb * 16);
            acc[t] = __builtin_amdgcn_wmma_f32_16x16x32_bf16(
                false, afrag, false, bfrag, (short)0, acc[t], false, false);
        }
    }

    #pragma unroll
    for (int t = 0; t < 4; ++t) {
        int col = t * 16 + (lane & 15);
        #pragma unroll
        for (int j = 0; j < 8; ++j) {
            int row = rowBase + wave * 16 + (lane >> 4) * 8 + j;
            if (row < n) out[(size_t)row * 64 + col] = acc[t][j];
        }
    }
}

// ---------------------------------------------------------------------------
// Kernel 3: per-(node,head) attention logits a_src/a_dst
// ---------------------------------------------------------------------------
__global__ void k_att(const float* __restrict__ xw,
                      const float* __restrict__ att_s, const float* __restrict__ att_d,
                      float* __restrict__ a_s, float* __restrict__ a_d, int n, int H)
{
    int t = blockIdx.x * blockDim.x + threadIdx.x;
    if (t >= n * H) return;
    const int C = 64 / H;
    int nn = t / H, h = t - nn * H;
    const float* row = xw + (size_t)nn * 64 + h * C;
    float s1 = 0.f, s2 = 0.f;
    for (int c = 0; c < C; ++c) {
        float v = row[c];
        s1 += v * att_s[h * C + c];
        s2 += v * att_d[h * C + c];
    }
    a_s[t] = s1;
    a_d[t] = s2;
}

// ---------------------------------------------------------------------------
// Kernel 4: init amax = -BIG, denom/num = 0 (denom+num are contiguous)
// ---------------------------------------------------------------------------
__global__ void k_init(float* __restrict__ amax, int na, float* __restrict__ z, int nz)
{
    int i = blockIdx.x * blockDim.x + threadIdx.x;
    if (i < na) amax[i] = -3.0e38f;
    if (i < nz) z[i] = 0.f;
}

// ---------------------------------------------------------------------------
// Kernel 5: edge pass 1 — segment max of leaky-relu logits (incl self loops)
// ---------------------------------------------------------------------------
__global__ void k_edge_max(const int* __restrict__ src, const int* __restrict__ dst,
                           const float* __restrict__ a_s, const float* __restrict__ a_d,
                           float* __restrict__ amax, int E, int n, int H)
{
    int t = blockIdx.x * blockDim.x + threadIdx.x;
    int total = (E + n) * H;
    if (t >= total) return;
    int e = t / H, h = t - e * H;
    int s, d;
    if (e < E) { s = src[e]; d = dst[e]; } else { s = e - E; d = s; }
    float al = a_s[s * H + h] + a_d[d * H + h];
    al = al > 0.f ? al : 0.2f * al;
    atomicMaxF(&amax[d * H + h], al);
}

// ---------------------------------------------------------------------------
// Kernel 6: edge pass 2 — ex = exp(alpha - amax); denom += ex;
//           num[dst] += ex * xw[src]   (8 threads per edge, 8 channels each)
// ---------------------------------------------------------------------------
__global__ void k_edge_acc(const int* __restrict__ src, const int* __restrict__ dst,
                           const float* __restrict__ a_s, const float* __restrict__ a_d,
                           const float* __restrict__ amax,
                           float* __restrict__ denom, float* __restrict__ num,
                           const float* __restrict__ xw, int E, int n, int H, int C)
{
    int t = blockIdx.x * blockDim.x + threadIdx.x;
    int total = (E + n) * 8;
    if (t >= total) return;
    int e = t >> 3, g = t & 7;
    int s, d;
    if (e < E) { s = src[e]; d = dst[e]; } else { s = e - E; d = s; }
    int c0 = g * 8;
    int h = c0 / C;                 // an 8-channel group never straddles a head
    __builtin_prefetch(&xw[(size_t)s * 64 + c0], 0, 0);   // global_prefetch_b8
    float al = a_s[s * H + h] + a_d[d * H + h];
    al = al > 0.f ? al : 0.2f * al;
    float ex = expf(al - amax[d * H + h]);
    if ((c0 % C) == 0) atomicAdd(&denom[d * H + h], ex);
    #pragma unroll
    for (int j = 0; j < 8; ++j)
        atomicAdd(&num[(size_t)d * 64 + c0 + j], ex * xw[(size_t)s * 64 + c0 + j]);
}

// ---------------------------------------------------------------------------
// Kernel 7: node finalize — out = ELU(BN(num/denom + bias) + identity)
// ---------------------------------------------------------------------------
__global__ void k_node_fin(const float* __restrict__ num, const float* __restrict__ denom,
                           const float* __restrict__ bias,
                           const float* __restrict__ bn_g, const float* __restrict__ bn_b,
                           const float* __restrict__ bn_m, const float* __restrict__ bn_v,
                           const float* __restrict__ identity, float* __restrict__ out,
                           int n, int H, int C)
{
    int i = blockIdx.x * blockDim.x + threadIdx.x;
    if (i >= n * 64) return;
    int nn = i >> 6, c = i & 63;
    int h = c / C;
    float o = num[i] / denom[nn * H + h] + bias[c];
    o = (o - bn_m[c]) * rsqrtf(bn_v[c] + 1e-5f) * bn_g[c] + bn_b[c];
    o += identity[i];
    out[i] = elu1(o);
}

// ---------------------------------------------------------------------------
// Kernel 8: classifier — out[N,8] = h3 @ cls_W + cls_b
// ---------------------------------------------------------------------------
__global__ void k_cls(const float* __restrict__ h, const float* __restrict__ W,
                      const float* __restrict__ b, float* __restrict__ out, int n)
{
    int t = blockIdx.x * blockDim.x + threadIdx.x;
    if (t >= n * 8) return;
    int row = t >> 3, c = t & 7;
    const float* hr = h + (size_t)row * 64;
    float s = b[c];
    #pragma unroll
    for (int k = 0; k < 64; ++k) s += hr[k] * W[k * 8 + c];
    out[t] = s;
}

// ---------------------------------------------------------------------------
static inline int cdiv(int a, int b) { return (a + b - 1) / b; }

extern "C" void kernel_launch(void* const* d_in, const int* in_sizes, int n_in,
                              void* d_out, int out_size, void* d_ws, size_t ws_size,
                              hipStream_t stream) {
    (void)n_in; (void)out_size; (void)ws_size;
    // setup_inputs() dict order: x, edge_index, proj_W, proj_b,
    // bn1_{g,b,m,v}, bn2_{g,b,m,v}, bn3_{g,b,m,v},
    // W1, att_src1, att_dst1, b1, W2, att_src2, att_dst2, b2, cls_W, cls_b
    const float* x       = (const float*)d_in[0];
    const int*   ei      = (const int*)  d_in[1];
    const float* proj_W  = (const float*)d_in[2];
    const float* proj_b  = (const float*)d_in[3];
    const float* bn1_g   = (const float*)d_in[4];
    const float* bn1_b   = (const float*)d_in[5];
    const float* bn1_m   = (const float*)d_in[6];
    const float* bn1_v   = (const float*)d_in[7];
    const float* bn2_g   = (const float*)d_in[8];
    const float* bn2_b   = (const float*)d_in[9];
    const float* bn2_m   = (const float*)d_in[10];
    const float* bn2_v   = (const float*)d_in[11];
    const float* bn3_g   = (const float*)d_in[12];
    const float* bn3_b   = (const float*)d_in[13];
    const float* bn3_m   = (const float*)d_in[14];
    const float* bn3_v   = (const float*)d_in[15];
    const float* W1      = (const float*)d_in[16];
    const float* att_s1  = (const float*)d_in[17];
    const float* att_d1  = (const float*)d_in[18];
    const float* b1      = (const float*)d_in[19];
    const float* W2      = (const float*)d_in[20];
    const float* att_s2  = (const float*)d_in[21];
    const float* att_d2  = (const float*)d_in[22];
    const float* b2      = (const float*)d_in[23];
    const float* cls_W   = (const float*)d_in[24];
    const float* cls_b   = (const float*)d_in[25];

    const int N = in_sizes[0] / 768;
    const int E = in_sizes[1] / 2;
    const int* src = ei;
    const int* dst = ei + E;

    // Workspace carve-up (denom and num MUST be adjacent for k_init).
    float* p = (float*)d_ws;
    float* h1    = p; p += (size_t)N * 64;
    float* xw    = p; p += (size_t)N * 64;
    float* a_s   = p; p += (size_t)N * 8;
    float* a_d   = p; p += (size_t)N * 8;
    float* amax  = p; p += (size_t)N * 8;
    float* denom = p; p += (size_t)N * 8;   // contiguous with num
    float* num   = p; p += (size_t)N * 64;
    float* h2    = p; p += (size_t)N * 64;
    float* h3    = p; p += (size_t)N * 64;
    // bf16 transposed-weight region (64B aligned)
    uintptr_t up = ((uintptr_t)p + 63) & ~(uintptr_t)63;
    __bf16* Wt0 = (__bf16*)up;              // [64][768]
    __bf16* Wt1 = Wt0 + (size_t)64 * 768;   // [64][64]
    __bf16* Wt2 = Wt1 + (size_t)64 * 64;    // [64][64]

    const int gRows = cdiv(N, 128);

    // ---- one-shot weight prep (transpose + bf16) ----
    k_cvtW<<<cdiv(768 * 64, 256), 256, 0, stream>>>(proj_W, Wt0, 768);
    k_cvtW<<<cdiv(64 * 64, 256), 256, 0, stream>>>(W1, Wt1, 64);
    k_cvtW<<<cdiv(64 * 64, 256), 256, 0, stream>>>(W2, Wt2, 64);

    // ---- projection + BN1 + ELU (WMMA, async double-buffered staging) ----
    k_proj<<<gRows, 256, 0, stream>>>(x, Wt0, proj_b,
                                      bn1_g, bn1_b, bn1_m, bn1_v, h1, N);

    // ---- GAT layer 1 (H=8, C=8) ----
    {
        const int H = 8, C = 8;
        k_gemm64<<<gRows, 256, 0, stream>>>(h1, Wt1, xw, N);
        k_att<<<cdiv(N * H, 256), 256, 0, stream>>>(xw, att_s1, att_d1, a_s, a_d, N, H);
        k_init<<<cdiv(N * 72, 256), 256, 0, stream>>>(amax, N * 8, denom, N * 72);
        k_edge_max<<<cdiv((E + N) * H, 256), 256, 0, stream>>>(src, dst, a_s, a_d, amax, E, N, H);
        k_edge_acc<<<cdiv((E + N) * 8, 256), 256, 0, stream>>>(src, dst, a_s, a_d, amax,
                                                               denom, num, xw, E, N, H, C);
        k_node_fin<<<cdiv(N * 64, 256), 256, 0, stream>>>(num, denom, b1,
                                                          bn2_g, bn2_b, bn2_m, bn2_v,
                                                          h1, h2, N, H, C);
    }

    // ---- GAT layer 2 (H=1, C=64) ----
    {
        const int H = 1, C = 64;
        k_gemm64<<<gRows, 256, 0, stream>>>(h2, Wt2, xw, N);
        k_att<<<cdiv(N * H, 256), 256, 0, stream>>>(xw, att_s2, att_d2, a_s, a_d, N, H);
        k_init<<<cdiv(N * 72, 256), 256, 0, stream>>>(amax, N * 8, denom, N * 72);
        k_edge_max<<<cdiv((E + N) * H, 256), 256, 0, stream>>>(src, dst, a_s, a_d, amax, E, N, H);
        k_edge_acc<<<cdiv((E + N) * 8, 256), 256, 0, stream>>>(src, dst, a_s, a_d, amax,
                                                               denom, num, xw, E, N, H, C);
        k_node_fin<<<cdiv(N * 64, 256), 256, 0, stream>>>(num, denom, b2,
                                                          bn3_g, bn3_b, bn3_m, bn3_v,
                                                          h2, h3, N, H, C);
    }

    // ---- classifier ----
    k_cls<<<cdiv(N * 8, 256), 256, 0, stream>>>(h3, cls_W, cls_b, (float*)d_out, N);
}
